// MoEBlock_7181185319326
// MI455X (gfx1250) — matrix operand order repaired and checked by reading, not compiled, not run on hardware
//
#include <hip/hip_runtime.h>
#include <hip/hip_bf16.h>
#include <math.h>

typedef __attribute__((ext_vector_type(16))) _Float16 v16h;
typedef __attribute__((ext_vector_type(8)))  _Float16 v8h;
typedef __attribute__((ext_vector_type(8)))  float    v8f;

#define BB   16
#define CC   64
#define EE   4
#define LL   5
#define HH   160
#define WWD  160
#define HWs  (HH*WWD)

__device__ __forceinline__ float gelu_exact(float v) {
    return 0.5f * v * (1.0f + erff(v * 0.70710678118654752f));
}

// ---------------------------------------------------------------------------
// Weight packing: f16, pre-laid-out in WMMA B-fragment order so each lane
// loads its 16 halves as two contiguous b128s.
//   WB[((nt*Kch + kc)*32 + lane)*16 + i] = W[K=kc*32+(lane>>4)*8+(i&7)+(i>>3)*16][oc=nt*16+(lane&15)]
// ---------------------------------------------------------------------------
// c1_w1 (oc,ic,3,3), K = tap*64 + ic (tap = dy*3+dx), Kch=18, Ntiles=4
__global__ __launch_bounds__(256) void pack_c1w1_frag(const float* __restrict__ w,
                                                      _Float16* __restrict__ WB) {
    int idx = blockIdx.x * 256 + threadIdx.x;
    if (idx >= 576 * CC) return;
    int i    = idx & 15;
    int lane = (idx >> 4) & 31;
    int kc   = (idx >> 9) % 18;
    int nt   = idx / (512 * 18);
    int K  = kc * 32 + (lane >> 4) * 8 + (i & 7) + (i >> 3) * 16;
    int oc = nt * 16 + (lane & 15);
    int tap = K / 64, ic = K % 64;
    WB[idx] = (_Float16)w[((oc * CC + ic) * 3 + tap / 3) * 3 + tap % 3];
}
// generic 1x1 weights (Cout,Cin,1,1): K = ic
__global__ __launch_bounds__(256) void pack_1x1_frag(const float* __restrict__ w,
                                                     _Float16* __restrict__ WB,
                                                     int Cout, int Cin) {
    int idx = blockIdx.x * 256 + threadIdx.x;
    if (idx >= Cout * Cin) return;
    int Kch  = Cin >> 5;
    int i    = idx & 15;
    int lane = (idx >> 4) & 31;
    int kc   = (idx >> 9) % Kch;
    int nt   = idx / (512 * Kch);
    int ic = kc * 32 + (lane >> 4) * 8 + (i & 7) + (i >> 3) * 16;
    int oc = nt * 16 + (lane & 15);
    WB[idx] = (_Float16)w[oc * Cin + ic];
}

// ---------------------------------------------------------------------------
// conv 3x3 (64->64, pad 1) + exact gelu, implicit GEMM via WMMA f16->f32
// A = input patches (M=16 pixels, K=tap*64+ic) from LDS; B = frag-packed weights
// grid: (W/16, H, B), block 128 (4 waves = 4 oc tiles). Output NHWC f16.
// ---------------------------------------------------------------------------
__global__ __launch_bounds__(128) void conv3x3_gelu_wmma(const float* __restrict__ x,
                                                         const _Float16* __restrict__ WB,
                                                         const float* __restrict__ bias,
                                                         _Float16* __restrict__ h1t) {
    const int tx = blockIdx.x * 16;
    const int y  = blockIdx.y;
    const int b  = blockIdx.z;
    __shared__ __align__(16) _Float16 lds[3][18][CC];   // [row][col][channel]

    const int tid = threadIdx.x;
    for (int idx = tid; idx < 3 * 18 * CC; idx += 128) {
        int col = idx % 18;
        int r   = (idx / 18) % 3;
        int c   = idx / 54;
        int gy = y + r - 1;
        int gx = tx + col - 1;
        float v = 0.f;
        if (gy >= 0 && gy < HH && gx >= 0 && gx < WWD)
            v = x[((size_t)(b * CC + c) * HH + gy) * WWD + gx];
        lds[r][col][c] = (_Float16)v;
    }
    __syncthreads();

    const int wave = tid >> 5;        // oc tile 0..3
    const int lane = tid & 31;
    const int half = lane >> 4;
    const int ln   = lane & 15;
    const int oc_n = wave * 16 + ln;

    v8f acc = {};
#pragma unroll
    for (int kb = 0; kb < 576; kb += 32) {
        const int kc  = kb >> 5;
        const int tap = kb / 64;
        const int dy = tap / 3, dx = tap % 3;
        const int icb = kb % 64;
        // A: row M = ln (pixel tx+ln), two contiguous 16B LDS reads
        v8h alo = *(const v8h*)&lds[dy][ln + dx][icb + half * 8];
        v8h ahi = *(const v8h*)&lds[dy][ln + dx][icb + 16 + half * 8];
        v16h a = __builtin_shufflevector(alo, ahi, 0,1,2,3,4,5,6,7,8,9,10,11,12,13,14,15);
        // B: frag-packed weights, two contiguous 16B global reads (L2-resident)
        const v8h* wb = (const v8h*)(WB + (((size_t)wave * 18 + kc) * 32 + lane) * 16);
        v16h bf = __builtin_shufflevector(wb[0], wb[1], 0,1,2,3,4,5,6,7,8,9,10,11,12,13,14,15);
        acc = __builtin_amdgcn_wmma_f32_16x16x32_f16(false, a, false, bf,
                                                     (short)0, acc, false, false);
    }
#pragma unroll
    for (int r = 0; r < 8; ++r) {
        int pix = tx + half * 8 + r;
        float v = acc[r] + bias[oc_n];
        v = gelu_exact(v);
        h1t[((size_t)b * HWs + (size_t)y * WWD + pix) * CC + oc_n] = (_Float16)v;
    }
}

// ---------------------------------------------------------------------------
// Generic 1x1 conv GEMM via WMMA. NHWC f16 input (Cin=64, Kch=2).
// MODE 0: Cout=128, channel-shuffle split -> o0 (xs) / o1 (k), NHWC f16
// MODE 1: Cout=64, plain f32 store NCHW (proj -> d_out)
// grid: (B*HW/16, Cout/64), block 128
// ---------------------------------------------------------------------------
template <int MODE>
__global__ __launch_bounds__(128) void gemm1x1_wmma(const _Float16* __restrict__ in,
                                                    const _Float16* __restrict__ WB,
                                                    const float* __restrict__ bias,
                                                    _Float16* __restrict__ o0,
                                                    _Float16* __restrict__ o1,
                                                    float* __restrict__ of) {
    const size_t pix0 = (size_t)blockIdx.x * 16;     // global pixel over B*HW
    const int wave = threadIdx.x >> 5, lane = threadIdx.x & 31;
    const int half = lane >> 4, ln = lane & 15;
    const int mt = blockIdx.y * 4 + wave;
    const int oc_n = mt * 16 + ln;

    v8f acc = {};
#pragma unroll
    for (int kc = 0; kc < 2; ++kc) {
        const _Float16* arow = in + (pix0 + ln) * CC + kc * 32;
        v8h alo = *(const v8h*)(arow + half * 8);
        v8h ahi = *(const v8h*)(arow + 16 + half * 8);
        v16h a = __builtin_shufflevector(alo, ahi, 0,1,2,3,4,5,6,7,8,9,10,11,12,13,14,15);
        const v8h* wb = (const v8h*)(WB + (((size_t)mt * 2 + kc) * 32 + lane) * 16);
        v16h bf = __builtin_shufflevector(wb[0], wb[1], 0,1,2,3,4,5,6,7,8,9,10,11,12,13,14,15);
        acc = __builtin_amdgcn_wmma_f32_16x16x32_f16(false, a, false, bf,
                                                     (short)0, acc, false, false);
    }
#pragma unroll
    for (int r = 0; r < 8; ++r) {
        size_t pix = pix0 + half * 8 + r;
        float v = acc[r] + bias[oc_n];
        if (MODE == 0) {
            int m = 2 * (oc_n & 63) + (oc_n >> 6);   // channel shuffle (groups=2)
            _Float16* dst = (m < 64) ? o0 : o1;
            dst[pix * CC + (m & 63)] = (_Float16)v;
        } else {
            size_t b = pix / HWs, p = pix % HWs;
            of[(b * CC + oc_n) * HWs + p] = v;
        }
    }
}

// ---------------------------------------------------------------------------
// Fused depthwise 1x3 -> 3x1 + gelu (sc path), NHWC f16, channel-fastest tid
// ---------------------------------------------------------------------------
__global__ __launch_bounds__(256) void sc_fused_kernel(const _Float16* __restrict__ xs0,
                                                       const float* __restrict__ w1,
                                                       const float* __restrict__ b1,
                                                       const float* __restrict__ w2,
                                                       const float* __restrict__ b2,
                                                       _Float16* __restrict__ xsf) {
    size_t idx = (size_t)blockIdx.x * 256 + threadIdx.x;
    if (idx >= (size_t)BB * HWs * CC) return;
    int c = idx % CC;
    size_t pp = idx / CC;                 // global pixel
    int xw = pp % WWD;
    int y  = (pp / WWD) % HH;
    float acc = b2[c];
#pragma unroll
    for (int dy = 0; dy < 3; ++dy) {
        int yy = y + dy - 1;
        if (yy < 0 || yy >= HH) continue;
        float t = b1[c];
#pragma unroll
        for (int dx = 0; dx < 3; ++dx) {
            int xx = xw + dx - 1;
            if (xx < 0 || xx >= WWD) continue;
            t += w1[c * 3 + dx] * (float)xs0[(pp + (size_t)(dy - 1) * WWD + (dx - 1)) * CC + c];
        }
        acc += w2[c * 3 + dy] * t;
    }
    xsf[idx] = (_Float16)gelu_exact(acc);
}

// ---------------------------------------------------------------------------
// Depthwise 4x4 stride-4 pad-2 + gelu (agg), NHWC, channel-fastest tid
// ---------------------------------------------------------------------------
template <typename Tin>
__global__ __launch_bounds__(256) void agg_dw_kernel(const Tin* __restrict__ in,
                                                     const float* __restrict__ w,
                                                     const float* __restrict__ bias,
                                                     float* __restrict__ out,
                                                     int Hin, int Win, int Ho, int Wo) {
    int idx = blockIdx.x * 256 + threadIdx.x;
    int total = BB * Ho * Wo * CC;
    if (idx >= total) return;
    int c  = idx % CC;
    int po = idx / CC;
    int wo = po % Wo;
    int ho = (po / Wo) % Ho;
    int b  = po / (Wo * Ho);
    float acc = bias[c];
#pragma unroll
    for (int ky = 0; ky < 4; ++ky) {
        int yy = ho * 4 + ky - 2;
        if (yy < 0 || yy >= Hin) continue;
#pragma unroll
        for (int kx = 0; kx < 4; ++kx) {
            int xx = wo * 4 + kx - 2;
            if (xx < 0 || xx >= Win) continue;
            acc += w[c * 16 + ky * 4 + kx] *
                   (float)in[((size_t)b * Hin * Win + (size_t)yy * Win + xx) * CC + c];
        }
    }
    out[idx] = gelu_exact(acc);
}

// cal_dw: depthwise 3x3 pad 1 on 11x11 NHWC (no activation)
__global__ __launch_bounds__(256) void cal_dw_kernel(const float* __restrict__ in,
                                                     const float* __restrict__ w,
                                                     const float* __restrict__ bias,
                                                     float* __restrict__ out) {
    int idx = blockIdx.x * 256 + threadIdx.x;
    if (idx >= BB * 121 * CC) return;
    int c  = idx % CC;
    int po = idx / CC;
    int xw = po % 11;
    int y  = (po / 11) % 11;
    int b  = po / 121;
    float acc = bias[c];
#pragma unroll
    for (int dy = 0; dy < 3; ++dy) {
        int yy = y + dy - 1;
        if (yy < 0 || yy >= 11) continue;
#pragma unroll
        for (int dx = 0; dx < 3; ++dx) {
            int xx = xw + dx - 1;
            if (xx < 0 || xx >= 11) continue;
            acc += w[c * 9 + dy * 3 + dx] * in[((size_t)b * 121 + yy * 11 + xx) * CC + c];
        }
    }
    out[idx] = acc;
}

// cal_pw: 1x1 64->64 on 11x11 NHWC -- tiny, contiguous per thread
__global__ __launch_bounds__(256) void cal_pw_kernel(const float* __restrict__ in,
                                                     const float* __restrict__ w,
                                                     const float* __restrict__ bias,
                                                     float* __restrict__ out) {
    int idx = blockIdx.x * 256 + threadIdx.x;
    if (idx >= BB * 121 * CC) return;
    int oc = idx % CC;
    int po = idx / CC;
    float acc = bias[oc];
    const float* ib = in + (size_t)po * CC;
    const float* wr = w + oc * CC;
    for (int ic = 0; ic < CC; ++ic) acc += wr[ic] * ib[ic];
    out[idx] = acc;
}

// Mean over H,W of xs_final (NHWC) -> (B*C). Block per batch, lane = channel.
__global__ __launch_bounds__(256) void mean_kernel(const _Float16* __restrict__ xsf,
                                                   float* __restrict__ mean) {
    int b = blockIdx.x;
    int c = threadIdx.x & 63;
    int rep = threadIdx.x >> 6;          // 4 replicas per channel
    __shared__ float sm[256];
    float s = 0.f;
    for (int pix = rep; pix < HWs; pix += 4)
        s += (float)xsf[((size_t)b * HWs + pix) * CC + c];
    sm[threadIdx.x] = s;
    __syncthreads();
    if (threadIdx.x < 64) {
        float t = sm[threadIdx.x] + sm[threadIdx.x + 64] +
                  sm[threadIdx.x + 128] + sm[threadIdx.x + 192];
        mean[b * CC + threadIdx.x] = t / (float)HWs;
    }
}

// Gating: logits -> softmax -> top-2 sparse weights (B x E)
__global__ __launch_bounds__(32) void gate_kernel(const float* __restrict__ mean,
                                                  const float* __restrict__ gw,
                                                  float* __restrict__ expw) {
    int b = threadIdx.x;
    if (b >= BB) return;
    float lg[EE];
#pragma unroll
    for (int e = 0; e < EE; ++e) {
        float s = 0.f;
        for (int c = 0; c < CC; ++c) s += mean[b * CC + c] * gw[e * CC + c];
        lg[e] = s;
    }
    float m = lg[0];
#pragma unroll
    for (int e = 1; e < EE; ++e) m = fmaxf(m, lg[e]);
    float den = 0.f;
#pragma unroll
    for (int e = 0; e < EE; ++e) { lg[e] = expf(lg[e] - m); den += lg[e]; }
#pragma unroll
    for (int e = 0; e < EE; ++e) lg[e] /= den;
    int i0 = 0;
#pragma unroll
    for (int e = 1; e < EE; ++e) if (lg[e] > lg[i0]) i0 = e;   // first max (top_k tie rule)
    int i1 = -1;
#pragma unroll
    for (int e = 0; e < EE; ++e) {
        if (e == i0) continue;
        if (i1 < 0 || lg[e] > lg[i1]) i1 = e;
    }
#pragma unroll
    for (int e = 0; e < EE; ++e)
        expw[b * EE + e] = (e == i0 || e == i1) ? lg[e] : 0.f;
}

// Expert combine with inline bilinear 11->160 upsample of a4 added to k.
// All tensors NHWC; per-thread channel loops are contiguous 128B runs.
__global__ __launch_bounds__(256) void expert_kernel(const _Float16* __restrict__ xsf,
                                                     const _Float16* __restrict__ k0,
                                                     const float* __restrict__ a4,
                                                     const float* __restrict__ expw,
                                                     const float* __restrict__ ew1,
                                                     const float* __restrict__ eb1,
                                                     const float* __restrict__ ew2,
                                                     const float* __restrict__ eb2,
                                                     const float* __restrict__ ew3,
                                                     const float* __restrict__ eb3,
                                                     _Float16* __restrict__ outb) {
    int idx = blockIdx.x * 256 + threadIdx.x;
    if (idx >= BB * HWs) return;
    int b = idx / HWs;
    int p = idx % HWs;
    int y = p / WWD, xw = p % WWD;

    // half-pixel-center bilinear, edge clamp (jax.image.resize upsample)
    float fy = (y + 0.5f) * (11.0f / 160.0f) - 0.5f;
    float fx = (xw + 0.5f) * (11.0f / 160.0f) - 0.5f;
    int y0 = (int)floorf(fy); float ty = fy - (float)y0;
    int x0 = (int)floorf(fx); float tx = fx - (float)x0;
    int y0c = min(max(y0, 0), 10), y1c = min(max(y0 + 1, 0), 10);
    int x0c = min(max(x0, 0), 10), x1c = min(max(x0 + 1, 0), 10);
    float w00 = (1.f - ty) * (1.f - tx), w01 = (1.f - ty) * tx;
    float w10 = ty * (1.f - tx),         w11 = ty * tx;
    const float* q00 = a4 + ((size_t)b * 121 + y0c * 11 + x0c) * CC;
    const float* q01 = a4 + ((size_t)b * 121 + y0c * 11 + x1c) * CC;
    const float* q10 = a4 + ((size_t)b * 121 + y1c * 11 + x0c) * CC;
    const float* q11 = a4 + ((size_t)b * 121 + y1c * 11 + x1c) * CC;
    const _Float16* xrow = xsf + (size_t)idx * CC;
    const _Float16* krow = k0  + (size_t)idx * CC;

    float wE[EE];
#pragma unroll
    for (int e = 0; e < EE; ++e) wE[e] = expw[b * EE + e];

    float aacc[EE][LL], bacc[EE][LL];
#pragma unroll
    for (int e = 0; e < EE; ++e)
#pragma unroll
        for (int l = 0; l < LL; ++l) { aacc[e][l] = eb1[e * LL + l]; bacc[e][l] = eb2[e * LL + l]; }

    for (int c = 0; c < CC; ++c) {
        float xv = (float)xrow[c];
        float kv = (float)krow[c] + w00 * q00[c] + w01 * q01[c] + w10 * q10[c] + w11 * q11[c];
#pragma unroll
        for (int e = 0; e < EE; ++e)
#pragma unroll
            for (int l = 0; l < LL; ++l) {
                aacc[e][l] += ew1[(e * LL + l) * CC + c] * xv;
                bacc[e][l] += ew2[(e * LL + l) * CC + c] * kv;
            }
    }
    float pel[EE][LL];
#pragma unroll
    for (int e = 0; e < EE; ++e)
#pragma unroll
        for (int l = 0; l < LL; ++l) pel[e][l] = aacc[e][l] * bacc[e][l];

    _Float16* orow = outb + (size_t)idx * CC;
    for (int c = 0; c < CC; ++c) {
        float o = (float)xrow[c];
#pragma unroll
        for (int e = 0; e < EE; ++e) {
            float s = eb3[e * CC + c];
#pragma unroll
            for (int l = 0; l < LL; ++l) s += ew3[(e * CC + c) * LL + l] * pel[e][l];
            o += wE[e] * s;
        }
        orow[c] = (_Float16)o;
    }
}

// ---------------------------------------------------------------------------
extern "C" void kernel_launch(void* const* d_in, const int* in_sizes, int n_in,
                              void* d_out, int out_size, void* d_ws, size_t ws_size,
                              hipStream_t stream) {
    const float* x       = (const float*)d_in[0];
    const float* c1_w1   = (const float*)d_in[1];
    const float* c1_b1   = (const float*)d_in[2];
    const float* c1_w2   = (const float*)d_in[3];
    const float* c1_b2   = (const float*)d_in[4];
    const float* agg_w   = (const float*)d_in[5];
    const float* agg_b   = (const float*)d_in[6];
    const float* cal_dw_w = (const float*)d_in[7];
    const float* cal_dw_b = (const float*)d_in[8];
    const float* cal_pw_w = (const float*)d_in[9];
    const float* cal_pw_b = (const float*)d_in[10];
    const float* sc_w1   = (const float*)d_in[11];
    const float* sc_b1   = (const float*)d_in[12];
    const float* sc_w2   = (const float*)d_in[13];
    const float* sc_b2   = (const float*)d_in[14];
    const float* gate_w  = (const float*)d_in[15];
    const float* e_w1    = (const float*)d_in[16];
    const float* e_b1    = (const float*)d_in[17];
    const float* e_w2    = (const float*)d_in[18];
    const float* e_b2    = (const float*)d_in[19];
    const float* e_w3    = (const float*)d_in[20];
    const float* e_b3    = (const float*)d_in[21];
    const float* proj_w  = (const float*)d_in[22];
    const float* proj_b  = (const float*)d_in[23];
    float* out = (float*)d_out;

    char* base = (char*)d_ws;
    size_t off = 0;
    auto carve = [&](size_t bytes) -> char* {
        char* p = base + off;
        off = (off + bytes + 255) & ~(size_t)255;
        return p;
    };
    const size_t bigH = (size_t)BB * HWs * CC;       // 26.2M elems
    _Float16* WB1  = (_Float16*)carve(576 * CC * 2);
    _Float16* WB2  = (_Float16*)carve(CC * 2 * CC * 2);
    _Float16* WBp  = (_Float16*)carve(CC * CC * 2);
    _Float16* h1t  = (_Float16*)carve(bigH * 2);
    _Float16* xs0t = (_Float16*)carve(bigH * 2);
    _Float16* k0t  = (_Float16*)carve(bigH * 2);
    _Float16* xsft = (_Float16*)carve(bigH * 2);
    _Float16* outbt = (_Float16*)carve(bigH * 2);
    float* a1t  = (float*)carve((size_t)BB * 41 * 41 * CC * 4);
    float* a2t  = (float*)carve((size_t)BB * 121 * CC * 4);
    float* a3t  = (float*)carve((size_t)BB * 121 * CC * 4);
    float* a4t  = (float*)carve((size_t)BB * 121 * CC * 4);
    float* mean = (float*)carve(BB * CC * 4);
    float* expw = (float*)carve(BB * EE * 4);
    (void)ws_size; (void)n_in; (void)in_sizes; (void)out_size;

    // 1) weight packing (WMMA fragment order)
    pack_c1w1_frag<<<(576 * CC + 255) / 256, 256, 0, stream>>>(c1_w1, WB1);
    pack_1x1_frag<<<(2 * CC * CC + 255) / 256, 256, 0, stream>>>(c1_w2, WB2, 2 * CC, CC);
    pack_1x1_frag<<<(CC * CC + 255) / 256, 256, 0, stream>>>(proj_w, WBp, CC, CC);

    // 2) conv3x3 + gelu (WMMA) -> h1 NHWC f16
    conv3x3_gelu_wmma<<<dim3(WWD / 16, HH, BB), 128, 0, stream>>>(x, WB1, c1_b1, h1t);

    // 3) 1x1 64->128 + channel-shuffle split (WMMA) -> xs0/k0 NHWC f16
    gemm1x1_wmma<0><<<dim3(BB * HWs / 16, 2), 128, 0, stream>>>(h1t, WB2, c1_b2, xs0t, k0t, nullptr);

    // 4) xs path: fused depthwise 1x3 -> 3x1 + gelu
    {
        size_t n = (size_t)BB * HWs * CC;
        sc_fused_kernel<<<(unsigned)((n + 255) / 256), 256, 0, stream>>>(xs0t, sc_w1, sc_b1, sc_w2, sc_b2, xsft);
    }

    // 5) k path: two stride-4 4x4 depthwise + gelu; then cal_dw, cal_pw (NHWC)
    agg_dw_kernel<_Float16><<<(BB * 41 * 41 * CC + 255) / 256, 256, 0, stream>>>(k0t, agg_w, agg_b, a1t, HH, WWD, 41, 41);
    agg_dw_kernel<float><<<(BB * 121 * CC + 255) / 256, 256, 0, stream>>>(a1t, agg_w, agg_b, a2t, 41, 41, 11, 11);
    cal_dw_kernel<<<(BB * 121 * CC + 255) / 256, 256, 0, stream>>>(a2t, cal_dw_w, cal_dw_b, a3t);
    cal_pw_kernel<<<(BB * 121 * CC + 255) / 256, 256, 0, stream>>>(a3t, cal_pw_w, cal_pw_b, a4t);

    // 6) gating
    mean_kernel<<<BB, 256, 0, stream>>>(xsft, mean);
    gate_kernel<<<1, 32, 0, stream>>>(mean, gate_w, expw);

    // 7) experts (bilinear resize fused in) -> outb NHWC f16
    expert_kernel<<<(BB * HWs + 255) / 256, 256, 0, stream>>>(xsft, k0t, a4t, expw,
        e_w1, e_b1, e_w2, e_b2, e_w3, e_b3, outbt);

    // 8) proj 1x1 64->64 (WMMA) -> d_out NCHW f32
    gemm1x1_wmma<1><<<dim3(BB * HWs / 16, 1), 128, 0, stream>>>(outbt, WBp, proj_b, nullptr, nullptr, out);
}
